// DeepProSite_73143293051183
// MI455X (gfx1250) — compile-verified, matrix operand order checked by reading, not run on hardware
//
#include <hip/hip_runtime.h>
#include <math.h>

typedef _Float16 half_t;
typedef __attribute__((ext_vector_type(16))) _Float16 v16h;
typedef __attribute__((ext_vector_type(8)))  _Float16 v8h;
typedef __attribute__((ext_vector_type(8)))  float    v8f;

#define BB      4
#define LL      1024
#define NB      (BB*LL)        // 4096 nodes
#define KNN     30
#define NODE_F  1038
#define NODE_FP 1056           // 33*32, zero-padded K for node embedding
#define EDGE_F  16
#define HH      128
#define FF      512
#define HEADS   4
#define DHEAD   32

// ---------------------------------------------------------------------------
// WMMA helpers (wave32, 16x16x32 f16 -> f32). Layouts per CDNA5 ISA 7.12.2:
//  A (16x32): lane<16 -> row=lane, k in {0..7,16..23}; lane>=16 -> row=lane-16,
//             k in {8..15,24..31}.  B is read from pre-transposed BT[N][K]
//  with the same per-lane pattern (lane indexes N).  C/D: elem r ->
//  row = r + (lane>=16 ? 8 : 0), col = lane&15.
// ---------------------------------------------------------------------------
__device__ inline v8f wmma16(v16h a, v16h b, v8f c) {
  return __builtin_amdgcn_wmma_f32_16x16x32_f16(
      false, a, false, b, (short)0, c, false, false);
}

// Fragment loader: two 16B vector loads per lane (ld, offsets multiple of 8).
__device__ inline v16h frag8(const half_t* p0, int ld) {
  int lane = threadIdx.x & 31;
  const half_t* p = p0 + (lane & 15) * ld + ((lane & 16) ? 8 : 0);
  v8h lo = *(const v8h*)p;
  v8h hi = *(const v8h*)(p + 16);
  return __builtin_shufflevector(lo, hi, 0, 1, 2, 3, 4, 5, 6, 7,
                                 8, 9, 10, 11, 12, 13, 14, 15);
}

// f32 x f16 dot with vectorized 16B weight loads (n multiple of 8, w 16B-aligned)
__device__ inline float dot_h(const float* x, const half_t* w, int n) {
  float a = 0.f;
  for (int c = 0; c < n; c += 8) {
    v8h wv = *(const v8h*)(w + c);
#pragma unroll
    for (int e = 0; e < 8; ++e) a += x[c + e] * (float)wv[e];
  }
  return a;
}

// ---------------------------------------------------------------------------
// Weight convert + transpose with K padding:
//   W [Kd][Nd] f32  ->  WT [Nd][KdPad] f16, zero-padded k in [Kd, KdPad)
// ---------------------------------------------------------------------------
__global__ void wt_kernel(const float* __restrict__ W, half_t* __restrict__ WT,
                          int Kd, int Nd, int KdPad) {
  int idx = blockIdx.x * 256 + threadIdx.x;
  if (idx >= Nd * KdPad) return;
  int n = idx / KdPad, k = idx - n * KdPad;
  WT[idx] = (k < Kd) ? (half_t)W[k * Nd + n] : (half_t)0.f;
}

// V [NB][NODE_F] f32 -> Vh [NB][NODE_FP] f16 zero-padded
__global__ void vpad_kernel(const float* __restrict__ V,
                            half_t* __restrict__ Vh) {
  int idx = blockIdx.x * 256 + threadIdx.x;
  if (idx >= NB * NODE_FP) return;
  int r = idx / NODE_FP, k = idx - r * NODE_FP;
  Vh[idx] = (k < NODE_F) ? (half_t)V[(size_t)r * NODE_F + k] : (half_t)0.f;
}

// ---------------------------------------------------------------------------
// kNN: per node, masked distances over L, row-max adjust, 30x block argmin.
// E_idx stores GLOBAL node index (b*L + j).
// ---------------------------------------------------------------------------
__global__ __launch_bounds__(256) void knn_kernel(
    const float* __restrict__ X, const float* __restrict__ mask,
    int* __restrict__ E_idx, float* __restrict__ D_nb,
    float* __restrict__ mask_nb) {
  int node = blockIdx.x;
  int b = node >> 10;
  int tid = threadIdx.x;
  const float* Xb = X + (size_t)b * LL * 3;
  const float* mb = mask + (size_t)b * LL;
  float xi = Xb[(node & 1023) * 3 + 0];
  float yi = Xb[(node & 1023) * 3 + 1];
  float zi = Xb[(node & 1023) * 3 + 2];
  float mi = mask[node];

  __shared__ float s_D[LL];
  __shared__ float s_r[256];
  __shared__ int   s_i[256];

  float lmax = -1e30f;
  for (int j = tid; j < LL; j += 256) {
    float dx = xi - Xb[j * 3 + 0];
    float dy = yi - Xb[j * 3 + 1];
    float dz = zi - Xb[j * 3 + 2];
    float m2 = mi * mb[j];
    float d = m2 * sqrtf(dx * dx + dy * dy + dz * dz + 1e-6f);
    s_D[j] = d;
    lmax = fmaxf(lmax, d);
  }
  s_r[tid] = lmax; __syncthreads();
  for (int off = 128; off > 0; off >>= 1) {
    if (tid < off) s_r[tid] = fmaxf(s_r[tid], s_r[tid + off]);
    __syncthreads();
  }
  float rmax = s_r[0]; __syncthreads();
  for (int j = tid; j < LL; j += 256) {
    float m2 = mi * mb[j];
    s_D[j] += (1.f - m2) * rmax;
  }
  __syncthreads();

  for (int k = 0; k < KNN; ++k) {
    float bv = 1e30f; int bi = 0;
    for (int j = tid; j < LL; j += 256) {
      float v = s_D[j];
      if (v < bv) { bv = v; bi = j; }
    }
    s_r[tid] = bv; s_i[tid] = bi; __syncthreads();
    for (int off = 128; off > 0; off >>= 1) {
      if (tid < off) {
        float v2 = s_r[tid + off];
        if (v2 < s_r[tid] || (v2 == s_r[tid] && s_i[tid + off] < s_i[tid])) {
          s_r[tid] = v2; s_i[tid] = s_i[tid + off];
        }
      }
      __syncthreads();
    }
    if (tid == 0) {
      int j = s_i[0];
      E_idx[node * KNN + k]   = b * LL + j;
      D_nb[node * KNN + k]    = s_r[0];
      mask_nb[node * KNN + k] = mi * mb[j];
      s_D[j] = 2e30f;
    }
    __syncthreads();
  }
}

// ---------------------------------------------------------------------------
// Edge embed: RBF(D_nb) @ W_edge + b_edge  -> h_E f16 [node][k][128]
// ---------------------------------------------------------------------------
__global__ __launch_bounds__(128) void edge_kernel(
    const float* __restrict__ D_nb, const float* __restrict__ W_edge,
    const float* __restrict__ b_edge, half_t* __restrict__ hE16) {
  int node = blockIdx.x;
  int h = threadIdx.x;
  float we[EDGE_F];
#pragma unroll
  for (int f = 0; f < EDGE_F; ++f) we[f] = W_edge[f * HH + h];
  float be = b_edge[h];
  const float inv_sig = 1.f / 1.25f;      // sigma = 20/16
  for (int k = 0; k < KNN; ++k) {
    float D = D_nb[node * KNN + k];
    float a = be;
#pragma unroll
    for (int f = 0; f < EDGE_F; ++f) {
      float mu = (20.f / 15.f) * (float)f;
      float t = (D - mu) * inv_sig;
      a += expf(-t * t) * we[f];
    }
    hE16[(size_t)(node * KNN + k) * HH + h] = (half_t)a;
  }
}

// ---------------------------------------------------------------------------
// Node embedding GEMM: hV = Vh[4096,1056] @ WTn^T + b_node  (WMMA, no tail)
// Block = 256 threads = 8 waves; wave w owns rows [blk*128 + w*16, +16),
// all 8 N-tiles in accumulators.  Per chunk: cluster A + 8 B fragment loads
// first, then the 8 WMMAs, so WMMA i overlaps the loads for i+1..7 via
// partial s_wait_loadcnt.
// ---------------------------------------------------------------------------
__global__ __launch_bounds__(256) void node_embed_kernel(
    const half_t* __restrict__ Vh, const half_t* __restrict__ WTn,
    const float* __restrict__ b_node, float* __restrict__ hV32,
    half_t* __restrict__ hV16) {
  int wv = threadIdx.x >> 5;
  int lane = threadIdx.x & 31;
  int row0 = blockIdx.x * 128 + wv * 16;

  v8f acc[8];
#pragma unroll
  for (int nt = 0; nt < 8; ++nt) { v8f z = {}; acc[nt] = z; }

  for (int kc = 0; kc < NODE_FP; kc += 32) {
    v16h a = frag8(Vh + (size_t)row0 * NODE_FP + kc, NODE_FP);
    v16h b[8];
#pragma unroll
    for (int nt = 0; nt < 8; ++nt)
      b[nt] = frag8(WTn + (size_t)(nt * 16) * NODE_FP + kc, NODE_FP);
#pragma unroll
    for (int nt = 0; nt < 8; ++nt)
      acc[nt] = wmma16(a, b[nt], acc[nt]);
  }

  int rb = (lane & 16) ? 8 : 0;
#pragma unroll
  for (int nt = 0; nt < 8; ++nt) {
    int col = nt * 16 + (lane & 15);
    float bn = b_node[col];
#pragma unroll
    for (int r = 0; r < 8; ++r) {
      int row = row0 + r + rb;
      float v = acc[nt][r] + bn;
      hV32[(size_t)row * HH + col] = v;
      hV16[(size_t)row * HH + col] = (half_t)v;
    }
  }
}

// ---------------------------------------------------------------------------
// block reduce (128 threads)
// ---------------------------------------------------------------------------
__device__ inline float block_sum128(float v, float* s) {
  int tid = threadIdx.x;
  __syncthreads();            // protect s from previous use
  s[tid] = v; __syncthreads();
  for (int off = 64; off > 0; off >>= 1) {
    if (tid < off) s[tid] += s[tid + off];
    __syncthreads();
  }
  float r = s[0]; __syncthreads();
  return r;
}

// ---------------------------------------------------------------------------
// One full encoder layer, fused per node. 128 threads = 4 waves.
//  1) gather h_EV[32,256] f16 into LDS (rows >=30 zeroed), 16B chunks
//  2) WMMA: Kt = h_EV@Wk, Vt = h_EV@Wval   (32 tile-jobs over 4 waves);
//     per job: hoist the 8 global B fragments, then LDS-A + WMMA steps
//  3) VALU: Q, logits, masked softmax (4 heads x 30), weighted sum, Wo
//  4) LN1, FF (relu MLP 128->512->128), LN2, mask; write f32 + f16
// ---------------------------------------------------------------------------
__global__ __launch_bounds__(128) void layer_kernel(
    const float* __restrict__ hVi32, const half_t* __restrict__ hVi16,
    float* __restrict__ hVo32, half_t* __restrict__ hVo16,
    const half_t* __restrict__ hE16, const int* __restrict__ E_idx,
    const float* __restrict__ mask_nb, const float* __restrict__ mask,
    const half_t* __restrict__ WTq, const half_t* __restrict__ WTk,
    const half_t* __restrict__ WTv, const half_t* __restrict__ WTo,
    const half_t* __restrict__ WTf1, const half_t* __restrict__ WTf2,
    const float* __restrict__ bf1, const float* __restrict__ bf2,
    const float* __restrict__ g1, const float* __restrict__ be1,
    const float* __restrict__ g2, const float* __restrict__ be2) {
  int node = blockIdx.x;
  int tid = threadIdx.x;
  int lane = tid & 31;
  int wv = tid >> 5;

  __shared__ alignas(16) half_t s_hEV[32 * 264];
  __shared__ alignas(16) float  s_K[32 * 132];
  __shared__ alignas(16) float  s_V[32 * 132];
  __shared__ float s_q[HH], s_hc[HH], s_y[HH], s_att[128], s_red[HH];
  __shared__ float s_act[FF];

  s_hc[tid] = hVi32[(size_t)node * HH + tid];

  // gather h_EV = [ h_V[E_idx] | h_E ], 8 halfs per thread per step
  for (int idx = tid; idx < 32 * 32; idx += 128) {
    int r = idx >> 5;
    int c8 = (idx & 31) * 8;
    v8h v = {};
    if (r < KNN) {
      if (c8 < HH) v = *(const v8h*)(hVi16 +
                        (size_t)E_idx[node * KNN + r] * HH + c8);
      else         v = *(const v8h*)(hE16 +
                        (size_t)(node * KNN + r) * HH + (c8 - HH));
    }
    *(v8h*)(s_hEV + r * 264 + c8) = v;
  }
  __syncthreads();

  // Q projection (VALU): q[n] = sum_c hc[c] * WqT[n][c]
  s_q[tid] = dot_h(s_hc, WTq + (size_t)tid * HH, HH);

  // K / V projections via WMMA: 2 proj x 2 Mtiles x 8 Ntiles = 32 jobs
  for (int j = wv; j < 32; j += 4) {
    int proj = j >> 4, mt = (j >> 3) & 1, nt = j & 7;
    const half_t* BT = (proj ? WTv : WTk) + (size_t)(nt * 16) * 256;
    // hoist all global (long-latency) B fragments for this job
    v16h b[8];
#pragma unroll
    for (int kc = 0; kc < 8; ++kc) b[kc] = frag8(BT + kc * 32, 256);
    v8f acc = {};
#pragma unroll
    for (int kc = 0; kc < 8; ++kc) {
      v16h a = frag8(s_hEV + mt * 16 * 264 + kc * 32, 264);
      acc = wmma16(a, b[kc], acc);
    }
    float* dst = proj ? s_V : s_K;
    int rb = (lane & 16) ? 8 : 0;
    int col = nt * 16 + (lane & 15);
#pragma unroll
    for (int r = 0; r < 8; ++r) dst[(mt * 16 + r + rb) * 132 + col] = acc[r];
  }
  __syncthreads();

  // logits [head][k]
  const float scale = 0.17677669529663687f;   // 1/sqrt(32)
  if (tid < HEADS * KNN) {
    int h = tid / KNN, k = tid - h * KNN;
    float d = 0.f;
#pragma unroll
    for (int dd = 0; dd < DHEAD; ++dd)
      d += s_q[h * DHEAD + dd] * s_K[k * 132 + h * DHEAD + dd];
    d *= scale;
    if (mask_nb[node * KNN + k] <= 0.f) d = -1e30f;
    s_att[h * KNN + k] = d;
  }
  __syncthreads();

  // softmax per head (masked lanes -> 0 weight)
  if (tid < HEADS) {
    float mx = -1e30f;
    for (int k = 0; k < KNN; ++k) mx = fmaxf(mx, s_att[tid * KNN + k]);
    float s = 0.f;
    for (int k = 0; k < KNN; ++k) {
      float lg = s_att[tid * KNN + k];
      float e = (lg <= -1e29f) ? 0.f : expf(lg - mx);
      s_att[tid * KNN + k] = e; s += e;
    }
    float inv = (s > 0.f) ? 1.f / s : 0.f;
    for (int k = 0; k < KNN; ++k) s_att[tid * KNN + k] *= inv;
  }
  __syncthreads();

  // dh[h*32+d] = sum_k att[h][k] * Vt[k][h*32+d]
  {
    int h = tid >> 5;
    float d = 0.f;
#pragma unroll
    for (int k = 0; k < KNN; ++k) d += s_att[h * KNN + k] * s_V[k * 132 + tid];
    s_red[tid] = d;
  }
  __syncthreads();

  // output projection Wo + residual
  float o = dot_h(s_red, WTo + (size_t)tid * HH, HH);
  float res = s_hc[tid] + o;

  // LayerNorm 1
  {
    float mu = block_sum128(res, s_red) * (1.f / 128.f);
    float dv = res - mu;
    float var = block_sum128(dv * dv, s_red) * (1.f / 128.f);
    float yn = dv * rsqrtf(var + 1e-5f);
    s_y[tid] = yn * g1[tid] + be1[tid];
  }
  __syncthreads();

  // FF layer 1 (+relu)
  for (int n = tid; n < FF; n += 128) {
    float a = bf1[n] + dot_h(s_y, WTf1 + (size_t)n * HH, HH);
    s_act[n] = (a > 0.f) ? a : 0.f;
  }
  __syncthreads();

  // FF layer 2 + residual
  float o2 = bf2[tid] + dot_h(s_act, WTf2 + (size_t)tid * FF, FF);
  float res2 = s_y[tid] + o2;

  // LayerNorm 2 + mask
  {
    float mu = block_sum128(res2, s_red) * (1.f / 128.f);
    float dv = res2 - mu;
    float var = block_sum128(dv * dv, s_red) * (1.f / 128.f);
    float yn = dv * rsqrtf(var + 1e-5f);
    float hout = (yn * g2[tid] + be2[tid]) * mask[node];
    hVo32[(size_t)node * HH + tid] = hout;
    hVo16[(size_t)node * HH + tid] = (half_t)hout;
  }
}

// ---------------------------------------------------------------------------
// Final projection: out[node] = hV . W_out + b_out
// ---------------------------------------------------------------------------
__global__ __launch_bounds__(128) void out_kernel(
    const float* __restrict__ hV32, const float* __restrict__ W_out,
    const float* __restrict__ b_out, float* __restrict__ out) {
  int idx = blockIdx.x * 128 + threadIdx.x;
  const float* h = hV32 + (size_t)idx * HH;
  float a = b_out[0];
  for (int c = 0; c < HH; ++c) a += h[c] * W_out[c];
  out[idx] = a;
}

// ---------------------------------------------------------------------------
extern "C" void kernel_launch(void* const* d_in, const int* in_sizes, int n_in,
                              void* d_out, int out_size, void* d_ws,
                              size_t ws_size, hipStream_t stream) {
  (void)in_sizes; (void)n_in; (void)out_size; (void)ws_size;
  const float* X      = (const float*)d_in[0];
  const float* V      = (const float*)d_in[1];
  const float* mask   = (const float*)d_in[2];
  const float* W_node = (const float*)d_in[3];
  const float* b_node = (const float*)d_in[4];
  const float* W_edge = (const float*)d_in[5];
  const float* b_edge = (const float*)d_in[6];
  const float* Wq     = (const float*)d_in[7];
  const float* Wk     = (const float*)d_in[8];
  const float* Wv     = (const float*)d_in[9];
  const float* Wo     = (const float*)d_in[10];
  const float* Wff1   = (const float*)d_in[11];
  const float* bff1   = (const float*)d_in[12];
  const float* Wff2   = (const float*)d_in[13];
  const float* bff2   = (const float*)d_in[14];
  const float* g1     = (const float*)d_in[15];
  const float* be1    = (const float*)d_in[16];
  const float* g2     = (const float*)d_in[17];
  const float* be2    = (const float*)d_in[18];
  const float* W_out  = (const float*)d_in[19];
  const float* b_out  = (const float*)d_in[20];
  float* out = (float*)d_out;

  // workspace bump allocator
  char* base = (char*)d_ws;
  size_t off = 0;
  auto alloc = [&](size_t nbytes) -> void* {
    void* p = base + off;
    off += (nbytes + 255) & ~(size_t)255;
    return p;
  };
  int*    eidx  = (int*)   alloc((size_t)NB * KNN * 4);
  float*  dnb   = (float*) alloc((size_t)NB * KNN * 4);
  float*  mnb   = (float*) alloc((size_t)NB * KNN * 4);
  half_t* hE    = (half_t*)alloc((size_t)NB * KNN * HH * 2);
  float*  hVa32 = (float*) alloc((size_t)NB * HH * 4);
  float*  hVb32 = (float*) alloc((size_t)NB * HH * 4);
  half_t* hVa16 = (half_t*)alloc((size_t)NB * HH * 2);
  half_t* hVb16 = (half_t*)alloc((size_t)NB * HH * 2);
  half_t* Vh    = (half_t*)alloc((size_t)NB * NODE_FP * 2);
  half_t* WTn   = (half_t*)alloc((size_t)HH * NODE_FP * 2);
  half_t* WTq   = (half_t*)alloc((size_t)4 * HH * HH * 2);
  half_t* WTk   = (half_t*)alloc((size_t)4 * HH * 256 * 2);
  half_t* WTv   = (half_t*)alloc((size_t)4 * HH * 256 * 2);
  half_t* WTo   = (half_t*)alloc((size_t)4 * HH * HH * 2);
  half_t* WTf1  = (half_t*)alloc((size_t)4 * FF * HH * 2);
  half_t* WTf2  = (half_t*)alloc((size_t)4 * HH * FF * 2);

  auto wt = [&](const float* W, half_t* WT, int Kd, int Nd, int KdPad) {
    int n = Nd * KdPad;
    wt_kernel<<<(n + 255) / 256, 256, 0, stream>>>(W, WT, Kd, Nd, KdPad);
  };
  wt(W_node, WTn, NODE_F, HH, NODE_FP);
  for (int li = 0; li < 4; ++li) {
    wt(Wq   + (size_t)li * HH * HH,  WTq  + (size_t)li * HH * HH,  HH,  HH, HH);
    wt(Wk   + (size_t)li * 256 * HH, WTk  + (size_t)li * HH * 256, 256, HH, 256);
    wt(Wv   + (size_t)li * 256 * HH, WTv  + (size_t)li * HH * 256, 256, HH, 256);
    wt(Wo   + (size_t)li * HH * HH,  WTo  + (size_t)li * HH * HH,  HH,  HH, HH);
    wt(Wff1 + (size_t)li * HH * FF,  WTf1 + (size_t)li * FF * HH,  HH,  FF, HH);
    wt(Wff2 + (size_t)li * FF * HH,  WTf2 + (size_t)li * HH * FF,  FF,  HH, FF);
  }
  {
    int n = NB * NODE_FP;
    vpad_kernel<<<(n + 255) / 256, 256, 0, stream>>>(V, Vh);
  }

  knn_kernel<<<NB, 256, 0, stream>>>(X, mask, eidx, dnb, mnb);
  edge_kernel<<<NB, 128, 0, stream>>>(dnb, W_edge, b_edge, hE);
  node_embed_kernel<<<NB / 128, 256, 0, stream>>>(Vh, WTn, b_node, hVa32, hVa16);

  const float* in32 = hVa32; const half_t* in16 = hVa16;
  float* o32 = hVb32; half_t* o16 = hVb16;
  for (int li = 0; li < 4; ++li) {
    layer_kernel<<<NB, 128, 0, stream>>>(
        in32, in16, o32, o16, hE, eidx, mnb, mask,
        WTq  + (size_t)li * HH * HH,  WTk + (size_t)li * HH * 256,
        WTv  + (size_t)li * HH * 256, WTo + (size_t)li * HH * HH,
        WTf1 + (size_t)li * FF * HH,  WTf2 + (size_t)li * HH * FF,
        bff1 + (size_t)li * FF, bff2 + (size_t)li * HH,
        g1 + (size_t)li * HH, be1 + (size_t)li * HH,
        g2 + (size_t)li * HH, be2 + (size_t)li * HH);
    const float* t32 = in32; const half_t* t16 = in16;
    in32 = o32; in16 = o16;
    o32 = (float*)t32; o16 = (half_t*)t16;
  }

  out_kernel<<<NB / 128, 128, 0, stream>>>(in32, W_out, b_out, out);
}
